// HybridAttentionBlock2d_11510512353734
// MI455X (gfx1250) — matrix operand order repaired and checked by reading, not compiled, not run on hardware
//
#include <hip/hip_runtime.h>
#include <hip/hip_bf16.h>

// ---------------------------------------------------------------------------
// HybridAttentionBlock2d for MI455X (gfx1250), wave32 + WMMA f16->f32.
//
// B=4, C=64, H=W=256 (P=65536 pixels/batch), HEADS=4, HD=16, K=9 (3x3 dil=2),
// HID=256.  3 kernels, intermediates: qkv in f16 (96MB, L2-resident),
// x1 (post-attn residual) in f32 (64MB). All GEMMs: v_wmma_f32_16x16x32_f16.
// ---------------------------------------------------------------------------

typedef __attribute__((ext_vector_type(16))) _Float16 v16h;
typedef __attribute__((ext_vector_type(8)))  float    v8f;

#define PIX_PER_BATCH 65536   // H*W
#define TILE_PIX      128     // pixels per workgroup

__device__ __forceinline__ v8f wmma_f16(v16h a, v16h b, v8f c) {
  // D = A(16x32 f16) * B(32x16 f16) + C(16x16 f32)
  return __builtin_amdgcn_wmma_f32_16x16x32_f16(
      /*neg_a=*/false, a, /*neg_b=*/false, b,
      /*c_mod=*/(short)0, c, /*reuse_a=*/false, /*reuse_b=*/false);
}

// A-matrix 16x32 f16 fragment (ISA 7.12.2): lane m = lane&15 holds row M=m.
// element i -> K = k0 + (i>>3)*16 + half*8 + (i&7)
__device__ __forceinline__ v16h load_a_frag(const _Float16* w, int row,
                                            int row_stride, int k0, int half) {
  v16h a;
#pragma unroll
  for (int i = 0; i < 16; ++i) {
    int k = k0 + ((i >> 3) << 4) + (half << 3) + (i & 7);
    a[i] = w[row * row_stride + k];
  }
  return a;
}

// B-matrix 32x16 f16 fragment: lane n = lane&15 holds column N=n,
// element i -> K = k0 + (i>>3)*16 + half*8 + (i&7)
__device__ __forceinline__ v16h load_b_frag(const _Float16* m, int k0,
                                            int half, int col, int col_stride) {
  v16h b;
#pragma unroll
  for (int i = 0; i < 16; ++i) {
    int k = k0 + ((i >> 3) << 4) + (half << 3) + (i & 7);
    b[i] = m[k * col_stride + col];
  }
  return b;
}

__device__ __forceinline__ float gelu_exact(float v) {
  return 0.5f * v * (1.0f + erff(v * 0.70710678118654752f));
}

// ---------------------------------------------------------------------------
// Kernel 1: LN1 + QKV projection.  qkv stored f16, layout [B][192][P].
// GEMM: M=192 (weights, A), N=128 pixels (activations, B), K=64.
// LDS: xs f32[64][130] | mu/rs f32[128+128] | xh f16[64][130] | wq f16[192*64]
// ---------------------------------------------------------------------------
__global__ __launch_bounds__(256) void k_ln_qkv(
    const float* __restrict__ x, const float* __restrict__ g1,
    const float* __restrict__ be1, const float* __restrict__ w_qkv,
    const float* __restrict__ b_qkv, _Float16* __restrict__ qkvh) {
  extern __shared__ char smem[];
  float*    xs = (float*)smem;                       // 64*130 f32
  float*    mu = xs + 64 * 130;                      // 128
  float*    rs = mu + 128;                           // 128
  _Float16* xh = (_Float16*)(rs + 128);              // 64*130 f16
  _Float16* wq = xh + 64 * 130;                      // 192*64 f16

  const int tid = threadIdx.x;
  const long long p0 = (long long)blockIdx.x * TILE_PIX;
  const int b  = (int)(p0 >> 16);
  const int pb = (int)(p0 & (PIX_PER_BATCH - 1));
  const float* xblk = x + (long long)b * 64 * PIX_PER_BATCH + pb;

  // stage QKV weights as f16
  for (int i = tid; i < 192 * 64; i += 256) wq[i] = (_Float16)w_qkv[i];
  // load x tile [64 channels][128 pixels] (coalesced along pixels)
  for (int i = tid; i < 64 * TILE_PIX; i += 256) {
    int c = i >> 7, j = i & 127;
    xs[c * 130 + j] = xblk[(long long)c * PIX_PER_BATCH + j];
  }
  __syncthreads();

  // LayerNorm statistics over the 64 channels, one thread per pixel
  if (tid < TILE_PIX) {
    float s = 0.f, s2 = 0.f;
    for (int c = 0; c < 64; ++c) {
      float v = xs[c * 130 + tid];
      s += v; s2 += v * v;
    }
    float m   = s * (1.0f / 64.0f);
    float var = s2 * (1.0f / 64.0f) - m * m;
    mu[tid] = m;
    rs[tid] = rsqrtf(var + 1e-5f);
  }
  __syncthreads();

  // normalize + affine -> f16 activations
  for (int i = tid; i < 64 * TILE_PIX; i += 256) {
    int c = i >> 7, j = i & 127;
    xh[c * 130 + j] =
        (_Float16)((xs[c * 130 + j] - mu[j]) * rs[j] * g1[c] + be1[c]);
  }
  __syncthreads();

  // WMMA: each wave owns 16 pixels, iterates 12 output tiles (192 channels)
  const int wave = tid >> 5, lane = tid & 31;
  const int half = lane >> 4, n = lane & 15;
  const int col  = wave * 16 + n;
  v16h bf0 = load_b_frag(xh, 0,  half, col, 130);
  v16h bf1 = load_b_frag(xh, 32, half, col, 130);
  _Float16* outp = qkvh + (long long)b * 192 * PIX_PER_BATCH + pb + col;
  for (int t = 0; t < 12; ++t) {
    const int obase = t * 16;
    v8f acc;
#pragma unroll
    for (int r = 0; r < 8; ++r) acc[r] = b_qkv[obase + r + half * 8];
    v16h a0 = load_a_frag(wq, obase + n, 64, 0,  half);
    v16h a1 = load_a_frag(wq, obase + n, 64, 32, half);
    acc = wmma_f16(a0, bf0, acc);
    acc = wmma_f16(a1, bf1, acc);
#pragma unroll
    for (int r = 0; r < 8; ++r) {
      int o = obase + r + half * 8;
      outp[(long long)o * PIX_PER_BATCH] = (_Float16)acc[r];
    }
  }
}

// ---------------------------------------------------------------------------
// Kernel 2: neighborhood attention (9 neighbors, dil=2, zero-pad semantics:
// OOB neighbor contributes score 0 inside softmax, value 0) + out-proj WMMA
// + residual.  qkv gathers hit L2 (96MB < 192MB).
// LDS: ao f16[64][136] | wo f16[64*64]
// ---------------------------------------------------------------------------
__global__ __launch_bounds__(256) void k_attn(
    const float* __restrict__ x, const _Float16* __restrict__ qkvh,
    const float* __restrict__ w_out, const float* __restrict__ b_out,
    float* __restrict__ x1) {
  extern __shared__ char smem[];
  _Float16* ao = (_Float16*)smem;   // 64*136 attn output, [channel][pixel]
  _Float16* wo = ao + 64 * 136;     // 64*64 out-proj weights

  const int tid = threadIdx.x;
  const long long p0 = (long long)blockIdx.x * TILE_PIX;
  const int b  = (int)(p0 >> 16);
  const int pb = (int)(p0 & (PIX_PER_BATCH - 1));

  for (int i = tid; i < 64 * 64; i += 256) wo[i] = (_Float16)w_out[i];

  const _Float16* qb = qkvh + (long long)b * 192 * PIX_PER_BATCH;

  // 512 tasks (128 pixels x 4 heads), 2 per thread; consecutive lanes ->
  // consecutive pixels, same head/channel -> coalesced gathers.
  for (int it = 0; it < 2; ++it) {
    int task = tid + (it << 8);
    int head = task >> 7, j = task & 127;
    int p  = pb + j;
    int ph = p >> 8, pw = p & 255;

    const _Float16* qp = qb + (long long)(head * 16) * PIX_PER_BATCH + p;
    float qv[16];
#pragma unroll
    for (int d = 0; d < 16; ++d)
      qv[d] = (float)qp[(long long)d * PIX_PER_BATCH] * 0.25f;  // hd^-0.5

    float sc[9];
#pragma unroll
    for (int ni = 0; ni < 3; ++ni) {
#pragma unroll
      for (int nj = 0; nj < 3; ++nj) {
        int nh = ph + (ni - 1) * 2, nw = pw + (nj - 1) * 2;
        float s = 0.f;
        if (nh >= 0 && nh < 256 && nw >= 0 && nw < 256) {
          const _Float16* kp =
              qb + (long long)(64 + head * 16) * PIX_PER_BATCH + (nh << 8) + nw;
#pragma unroll
          for (int d = 0; d < 16; ++d)
            s += qv[d] * (float)kp[(long long)d * PIX_PER_BATCH];
        }
        sc[ni * 3 + nj] = s;  // zero-padded k => score exactly 0
      }
    }
    float mx = sc[0];
#pragma unroll
    for (int q2 = 1; q2 < 9; ++q2) mx = fmaxf(mx, sc[q2]);
    float Z = 0.f;
#pragma unroll
    for (int q2 = 0; q2 < 9; ++q2) { sc[q2] = expf(sc[q2] - mx); Z += sc[q2]; }
    float inv = 1.0f / Z;

    float outv[16];
#pragma unroll
    for (int d = 0; d < 16; ++d) outv[d] = 0.f;
#pragma unroll
    for (int ni = 0; ni < 3; ++ni) {
#pragma unroll
      for (int nj = 0; nj < 3; ++nj) {
        int nh = ph + (ni - 1) * 2, nw = pw + (nj - 1) * 2;
        if (nh >= 0 && nh < 256 && nw >= 0 && nw < 256) {
          float wgt = sc[ni * 3 + nj] * inv;
          const _Float16* vp =
              qb + (long long)(128 + head * 16) * PIX_PER_BATCH + (nh << 8) + nw;
#pragma unroll
          for (int d = 0; d < 16; ++d)
            outv[d] += wgt * (float)vp[(long long)d * PIX_PER_BATCH];
        }
      }
    }
#pragma unroll
    for (int d = 0; d < 16; ++d)
      ao[(head * 16 + d) * 136 + j] = (_Float16)outv[d];
  }
  __syncthreads();

  // out-projection (64x64) + bias + residual  -> x1
  const int wave = tid >> 5, lane = tid & 31;
  const int half = lane >> 4, n = lane & 15;
  const int col  = wave * 16 + n;
  v16h bf0 = load_b_frag(ao, 0,  half, col, 136);
  v16h bf1 = load_b_frag(ao, 32, half, col, 136);
  const float* xp = x  + (long long)b * 64 * PIX_PER_BATCH + pb + col;
  float*       op = x1 + (long long)b * 64 * PIX_PER_BATCH + pb + col;
  for (int t = 0; t < 4; ++t) {
    const int obase = t * 16;
    v8f acc;
#pragma unroll
    for (int r = 0; r < 8; ++r) acc[r] = b_out[obase + r + half * 8];
    v16h a0 = load_a_frag(wo, obase + n, 64, 0,  half);
    v16h a1 = load_a_frag(wo, obase + n, 64, 32, half);
    acc = wmma_f16(a0, bf0, acc);
    acc = wmma_f16(a1, bf1, acc);
#pragma unroll
    for (int r = 0; r < 8; ++r) {
      int c = obase + r + half * 8;
      op[(long long)c * PIX_PER_BATCH] =
          xp[(long long)c * PIX_PER_BATCH] + acc[r];
    }
  }
}

// ---------------------------------------------------------------------------
// Kernel 3: LN2 + full MLP (64->256 GELU -> 64 GELU) + residual, fully fused:
// both weight matrices f16-resident in LDS, hidden tile never leaves LDS.
// LDS: xs f32[64][130] | mu/rs | xh f16[64][130] | w1 f16[256*64]
//      | w2 f16[64*256] | h f16[256][136]   (~186 KB, fits 320KB WGP LDS)
// ---------------------------------------------------------------------------
__global__ __launch_bounds__(256) void k_mlp(
    const float* __restrict__ x1, const float* __restrict__ g2,
    const float* __restrict__ be2, const float* __restrict__ w1,
    const float* __restrict__ b1, const float* __restrict__ w2,
    const float* __restrict__ b2, float* __restrict__ out) {
  extern __shared__ char smem[];
  float*    xs  = (float*)smem;                  // 64*130 f32 (x1 tile)
  float*    mu  = xs + 64 * 130;                 // 128
  float*    rs  = mu + 128;                      // 128
  _Float16* xh  = (_Float16*)(rs + 128);         // 64*130
  _Float16* w1h = xh + 64 * 130;                 // 256*64
  _Float16* w2h = w1h + 256 * 64;                // 64*256
  _Float16* hh  = w2h + 64 * 256;                // 256*136 hidden tile

  const int tid = threadIdx.x;
  const long long p0 = (long long)blockIdx.x * TILE_PIX;
  const int b  = (int)(p0 >> 16);
  const int pb = (int)(p0 & (PIX_PER_BATCH - 1));
  const float* xblk = x1 + (long long)b * 64 * PIX_PER_BATCH + pb;

  for (int i = tid; i < 256 * 64; i += 256) w1h[i] = (_Float16)w1[i];
  for (int i = tid; i < 64 * 256; i += 256) w2h[i] = (_Float16)w2[i];
  for (int i = tid; i < 64 * TILE_PIX; i += 256) {
    int c = i >> 7, j = i & 127;
    xs[c * 130 + j] = xblk[(long long)c * PIX_PER_BATCH + j];
  }
  __syncthreads();

  if (tid < TILE_PIX) {
    float s = 0.f, s2 = 0.f;
    for (int c = 0; c < 64; ++c) {
      float v = xs[c * 130 + tid];
      s += v; s2 += v * v;
    }
    float m   = s * (1.0f / 64.0f);
    float var = s2 * (1.0f / 64.0f) - m * m;
    mu[tid] = m;
    rs[tid] = rsqrtf(var + 1e-5f);
  }
  __syncthreads();
  for (int i = tid; i < 64 * TILE_PIX; i += 256) {
    int c = i >> 7, j = i & 127;
    xh[c * 130 + j] =
        (_Float16)((xs[c * 130 + j] - mu[j]) * rs[j] * g2[c] + be2[c]);
  }
  __syncthreads();

  const int wave = tid >> 5, lane = tid & 31;
  const int half = lane >> 4, n = lane & 15;
  const int col  = wave * 16 + n;

  // GEMM1: 256 = 16 tiles of 16 outputs, K=64; GELU into LDS hidden tile
  {
    v16h bf0 = load_b_frag(xh, 0,  half, col, 130);
    v16h bf1 = load_b_frag(xh, 32, half, col, 130);
    for (int t = 0; t < 16; ++t) {
      const int obase = t * 16;
      v8f acc;
#pragma unroll
      for (int r = 0; r < 8; ++r) acc[r] = b1[obase + r + half * 8];
      v16h a0 = load_a_frag(w1h, obase + n, 64, 0,  half);
      v16h a1 = load_a_frag(w1h, obase + n, 64, 32, half);
      acc = wmma_f16(a0, bf0, acc);
      acc = wmma_f16(a1, bf1, acc);
#pragma unroll
      for (int r = 0; r < 8; ++r) {
        int o = obase + r + half * 8;
        hh[o * 136 + col] = (_Float16)gelu_exact(acc[r]);
      }
    }
  }
  __syncthreads();

  // GEMM2: 4 output tiles, K=256 (8 WMMA k-steps); GELU + residual -> out
  {
    v16h hf[8];
#pragma unroll
    for (int s = 0; s < 8; ++s) hf[s] = load_b_frag(hh, s * 32, half, col, 136);
    float* op = out + (long long)b * 64 * PIX_PER_BATCH + pb + col;
    for (int t = 0; t < 4; ++t) {
      const int obase = t * 16;
      v8f acc;
#pragma unroll
      for (int r = 0; r < 8; ++r) acc[r] = b2[obase + r + half * 8];
#pragma unroll
      for (int s = 0; s < 8; ++s) {
        v16h a = load_a_frag(w2h, obase + n, 256, s * 32, half);
        acc = wmma_f16(a, hf[s], acc);
      }
#pragma unroll
      for (int r = 0; r < 8; ++r) {
        int c = obase + r + half * 8;
        op[(long long)c * PIX_PER_BATCH] =
            xs[c * 130 + col] + gelu_exact(acc[r]);
      }
    }
  }
}

// ---------------------------------------------------------------------------
extern "C" void kernel_launch(void* const* d_in, const int* in_sizes, int n_in,
                              void* d_out, int out_size, void* d_ws,
                              size_t ws_size, hipStream_t stream) {
  (void)in_sizes; (void)n_in; (void)out_size; (void)ws_size;
  const float* x      = (const float*)d_in[0];
  const float* gamma1 = (const float*)d_in[1];
  const float* beta1  = (const float*)d_in[2];
  const float* gamma2 = (const float*)d_in[3];
  const float* beta2  = (const float*)d_in[4];
  const float* w_qkv  = (const float*)d_in[5];
  const float* b_qkv  = (const float*)d_in[6];
  const float* w_out  = (const float*)d_in[7];
  const float* b_out  = (const float*)d_in[8];
  const float* w1     = (const float*)d_in[9];
  const float* b1     = (const float*)d_in[10];
  const float* w2     = (const float*)d_in[11];
  const float* b2     = (const float*)d_in[12];
  float* out = (float*)d_out;

  // workspace: qkv f16 [4][192][65536] (96MB) then x1 f32 (64MB)
  _Float16* qkvh = (_Float16*)d_ws;
  float*    x1   = (float*)((char*)d_ws + (size_t)4 * 192 * 65536 * 2);

  constexpr int SMEM1 = 64 * 130 * 4 + 256 * 4 + 64 * 130 * 2 + 192 * 64 * 2;
  constexpr int SMEM2 = 64 * 136 * 2 + 64 * 64 * 2;
  constexpr int SMEM3 = 64 * 130 * 4 + 256 * 4 + 64 * 130 * 2 +
                        256 * 64 * 2 + 64 * 256 * 2 + 256 * 136 * 2;

  // opt-in to >64KB dynamic LDS (WGP has 320KB); idempotent, capture-safe
  hipFuncSetAttribute((const void*)k_ln_qkv,
                      hipFuncAttributeMaxDynamicSharedMemorySize, SMEM1);
  hipFuncSetAttribute((const void*)k_mlp,
                      hipFuncAttributeMaxDynamicSharedMemorySize, SMEM3);

  const int nblocks = (4 * 65536) / TILE_PIX;  // 2048
  k_ln_qkv<<<nblocks, 256, SMEM1, stream>>>(x, gamma1, beta1, w_qkv, b_qkv,
                                            qkvh);
  k_attn<<<nblocks, 256, SMEM2, stream>>>(x, qkvh, w_out, b_out, x1);
  k_mlp<<<nblocks, 256, SMEM3, stream>>>(x1, gamma2, beta2, w1, b1, w2, b2,
                                         out);
}
